// MultiHeadAttention_11974368821597
// MI455X (gfx1250) — matrix operand order repaired and checked by reading, not compiled, not run on hardware
//
#include <hip/hip_runtime.h>

#define B_  4
#define T_  2048
#define D_  1024
#define H_  16
#define HS_ 64
#define M_  (B_ * T_)   // 8192

typedef __attribute__((ext_vector_type(16))) __bf16    v16bf;
typedef __attribute__((ext_vector_type(8)))  float     v8f;
typedef __attribute__((ext_vector_type(4)))  unsigned  v4u;
typedef __attribute__((ext_vector_type(8)))  unsigned  v8u;

union FragBF { uint4 u[2]; v16bf v; };

__device__ __forceinline__ unsigned short f2bf(float f) {
  unsigned u = __float_as_uint(f);
  unsigned r = u + 0x7FFFu + ((u >> 16) & 1u);   // round-to-nearest-even
  return (unsigned short)(r >> 16);
}
__device__ __forceinline__ unsigned pack2(float a, float b) {
  return (unsigned)f2bf(a) | ((unsigned)f2bf(b) << 16);
}

// A-fragment (16 x 32, 16-bit). ISA layout: lanes 0-15 row=lane, K={0..7,16..23};
// lanes 16-31 row=lane-16, K={8..15,24..31}. Source row-major with `stride`.
__device__ __forceinline__ v16bf load_fragA(const unsigned short* base, int stride, int kofs) {
  int lane = threadIdx.x & 31;
  int idx  = lane & 15;
  int kb   = (lane < 16) ? 0 : 8;
  const unsigned short* p = base + idx * stride + kofs + kb;
  FragBF f;
  f.u[0] = *(const uint4*)(p);        // K kb..kb+7
  f.u[1] = *(const uint4*)(p + 16);   // K kb+16..kb+23
  return f.v;
}

// B-fragment (K=32 x N=16, 16-bit). ISA layout: lanes 0-15 hold K={0..7,16..23} of
// column n=lane; lanes 16-31 hold K={8..15,24..31} of column n=lane-16.
// `base` points at the "column-source" matrix: row n holds column n's K values.
__device__ __forceinline__ v16bf load_fragB(const unsigned short* base, int stride, int kofs) {
  int lane = threadIdx.x & 31;
  int idx  = lane & 15;
  int kb   = (lane < 16) ? 0 : 8;
  const unsigned short* p = base + idx * stride + kofs + kb;
  FragBF f;
  f.u[0] = *(const uint4*)(p);        // K kb..kb+7
  f.u[1] = *(const uint4*)(p + 16);   // K kb+16..kb+23
  return f.v;
}

__device__ __forceinline__ v8f wmma_bf16(v16bf a, v16bf b, v8f c) {
  return __builtin_amdgcn_wmma_f32_16x16x32_bf16(false, a, false, b, (short)0, c, false, false);
}

// ---------------------------------------------------------------------------
// TDM: issue a 2D tensor_load_to_lds of bf16 (data_size = 2 bytes).
// Descriptor bit layout per CDNA5 ISA D# Group0 / Group1 (CDNA5 ISA).
// Groups 2/3 passed as zeroed SGPR quads (tile_dim3=0 => unused; 2D tile).
// ---------------------------------------------------------------------------
__device__ __forceinline__ void tdm_load_2d_bf16(unsigned lds_byte, unsigned long long ga,
                                                 unsigned tensor_d0, unsigned tensor_d1,
                                                 unsigned tile_d0, unsigned tile_d1,
                                                 unsigned stride0) {
  v4u g0;
  g0[0] = 1u;                                                    // count=1, user mode
  g0[1] = lds_byte;                                              // lds_addr
  g0[2] = (unsigned)ga;                                          // global_addr[31:0]
  g0[3] = (unsigned)((ga >> 32) & 0x01FFFFFFull) | (2u << 30);   // ga[56:32] | type=2
  v8u g1;
  g1[0] = (1u << 16);                                            // data_size=1 (2B), mask=0
  g1[1] = (tensor_d0 & 0xFFFFu) << 16;                           // tensor_dim0[15:0]
  g1[2] = (tensor_d0 >> 16) | ((tensor_d1 & 0xFFFFu) << 16);     // dim0[31:16] | dim1[15:0]
  g1[3] = (tensor_d1 >> 16) | (tile_d0 << 16);                   // dim1[31:16] | tile_dim0
  g1[4] = tile_d1;                                               // tile_dim1 (tile_dim2=0)
  g1[5] = stride0;                                               // tensor_dim0_stride[31:0]
  g1[6] = 0u;                                                    // stride0[47:32] | dim1_stride lo
  g1[7] = 0u;
  v4u gz = {0u, 0u, 0u, 0u};
  asm volatile("tensor_load_to_lds %0, %1, %2, %3"
               :: "s"(g0), "s"(g1), "s"(gz), "s"(gz)
               : "memory");
}

// ---------------------------------------------------------------------------
// Generic projection GEMM: Out[m][n] = sum_k A[m][k] * W[n][k]  (+ bias)
// MODE 0: A=f32 x,  store bf16 Q (scaled 1/8) as [B,H,T,HS]
// MODE 1: A=f32 x,  store bf16 K          as [B,H,T,HS]
// MODE 2: A=f32 x,  store bf16 V^T        as [B,H,HS,T]
// MODE 3: A=bf16 O, store f32 out + bias  as [M,D]
// ---------------------------------------------------------------------------
template<int MODE>
__global__ __launch_bounds__(128) void proj_kernel(const void* __restrict__ Ap,
                                                   const float* __restrict__ W,
                                                   const float* __restrict__ bias,
                                                   void* __restrict__ Out) {
  __shared__ unsigned short As[64 * 32];
  __shared__ unsigned short Bs[64 * 32];

  const int tid  = threadIdx.x;
  const int lane = tid & 31;
  const int wave = tid >> 5;
  const int m0 = blockIdx.y * 64;
  const int n0 = blockIdx.x * 64;
  const int wm = (wave >> 1) * 32;
  const int wn = (wave & 1) * 32;

  const int srow = tid >> 1;          // 0..63
  const int scol = (tid & 1) * 16;    // 0 or 16

  v8f acc[2][2] = {};

  for (int k0 = 0; k0 < D_; k0 += 32) {
    // ---- stage A tile (64 x 32) into LDS as bf16 ----
    if (MODE == 3) {
      const unsigned short* A = (const unsigned short*)Ap + (size_t)(m0 + srow) * D_ + k0 + scol;
      *(uint4*)(As + srow * 32 + scol)     = ((const uint4*)A)[0];
      *(uint4*)(As + srow * 32 + scol + 8) = ((const uint4*)A)[1];
    } else {
      const float* A = (const float*)Ap + (size_t)(m0 + srow) * D_ + k0 + scol;
      float4 a0 = ((const float4*)A)[0];
      float4 a1 = ((const float4*)A)[1];
      float4 a2 = ((const float4*)A)[2];
      float4 a3 = ((const float4*)A)[3];
      uint4 p0 = { pack2(a0.x, a0.y), pack2(a0.z, a0.w), pack2(a1.x, a1.y), pack2(a1.z, a1.w) };
      uint4 p1 = { pack2(a2.x, a2.y), pack2(a2.z, a2.w), pack2(a3.x, a3.y), pack2(a3.z, a3.w) };
      *(uint4*)(As + srow * 32 + scol)     = p0;
      *(uint4*)(As + srow * 32 + scol + 8) = p1;
    }
    // ---- stage W tile (rows n0..n0+63, cols k0..k0+31) as bf16 ----
    {
      const float* Wr = W + (size_t)(n0 + srow) * D_ + k0 + scol;
      float4 b0 = ((const float4*)Wr)[0];
      float4 b1 = ((const float4*)Wr)[1];
      float4 b2 = ((const float4*)Wr)[2];
      float4 b3 = ((const float4*)Wr)[3];
      uint4 q0 = { pack2(b0.x, b0.y), pack2(b0.z, b0.w), pack2(b1.x, b1.y), pack2(b1.z, b1.w) };
      uint4 q1 = { pack2(b2.x, b2.y), pack2(b2.z, b2.w), pack2(b3.x, b3.y), pack2(b3.z, b3.w) };
      *(uint4*)(Bs + srow * 32 + scol)     = q0;
      *(uint4*)(Bs + srow * 32 + scol + 8) = q1;
    }
    if (k0 + 32 < D_) {   // prefetch next K tile (-> global_prefetch_b8)
      if (MODE == 3)
        __builtin_prefetch((const unsigned short*)Ap + (size_t)(m0 + srow) * D_ + k0 + 32 + scol, 0, 3);
      else
        __builtin_prefetch((const float*)Ap + (size_t)(m0 + srow) * D_ + k0 + 32 + scol, 0, 3);
      __builtin_prefetch(W + (size_t)(n0 + srow) * D_ + k0 + 32 + scol, 0, 3);
    }
    __syncthreads();

    v16bf af0 = load_fragA(As + (wm + 0)  * 32, 32, 0);
    v16bf af1 = load_fragA(As + (wm + 16) * 32, 32, 0);
    v16bf bf0 = load_fragB(Bs + (wn + 0)  * 32, 32, 0);
    v16bf bf1 = load_fragB(Bs + (wn + 16) * 32, 32, 0);
    acc[0][0] = wmma_bf16(af0, bf0, acc[0][0]);
    acc[0][1] = wmma_bf16(af0, bf1, acc[0][1]);
    acc[1][0] = wmma_bf16(af1, bf0, acc[1][0]);
    acc[1][1] = wmma_bf16(af1, bf1, acc[1][1]);
    __syncthreads();
  }

  // ---- store; C layout: VGPR i -> row i (lanes0-15) / i+8 (lanes16-31), col = lane&15
  const int cn   = lane & 15;
  const int hi16 = lane >> 4;
  #pragma unroll
  for (int mi = 0; mi < 2; ++mi)
    #pragma unroll
    for (int ni = 0; ni < 2; ++ni)
      #pragma unroll
      for (int i = 0; i < 8; ++i) {
        int m = m0 + wm + mi * 16 + i + hi16 * 8;
        int n = n0 + wn + ni * 16 + cn;
        float v = acc[mi][ni][i];
        if (MODE == 3) {
          ((float*)Out)[(size_t)m * D_ + n] = v + bias[n];
        } else {
          int b = m >> 11, t = m & (T_ - 1);
          int h = n >> 6,  hs = n & (HS_ - 1);
          unsigned short bv = (MODE == 0) ? f2bf(v * 0.125f) : f2bf(v);
          if (MODE == 2)
            ((unsigned short*)Out)[((size_t)(b * H_ + h) * HS_ + hs) * T_ + t] = bv;   // V^T
          else
            ((unsigned short*)Out)[((size_t)(b * H_ + h) * T_ + t) * HS_ + hs] = bv;   // Q/K
        }
      }
}

// ---------------------------------------------------------------------------
// Flash attention: one block = 64 q-rows of one (b,h); 4 waves x 16 q-rows.
// Q pre-scaled by 1/sqrt(HS). kv tiles of 32 with online softmax.
// K / V^T tiles are DMA'd into LDS by the Tensor Data Mover (wave 0 issues,
// double-buffered, synced via s_wait_tensorcnt + workgroup barrier), removing
// the 4x-per-block redundant global reads of K/V.
// ---------------------------------------------------------------------------
__global__ __launch_bounds__(128) void attn_kernel(const unsigned short* __restrict__ qb,
                                                   const unsigned short* __restrict__ kb,
                                                   const unsigned short* __restrict__ vt,
                                                   unsigned short* __restrict__ ob) {
  __shared__ unsigned short Kt[2][32 * 64];    // kv-tile of K  (rows = kv, stride 64)
  __shared__ unsigned short Vt[2][64 * 32];    // kv-tile of V^T (rows = hs, stride 32)
  __shared__ unsigned short Pl[4 * 16 * 32];   // per-wave P staging (bf16)

  const int bh   = blockIdx.y;
  const int q0   = blockIdx.x * 64;
  const int tid  = threadIdx.x;
  const int lane = tid & 31;
  const int wave = tid >> 5;
  const int qw   = q0 + wave * 16;
  const int cn   = lane & 15;
  const int hi16 = lane >> 4;

  const unsigned short* qbase = qb + ((size_t)bh * T_ + qw) * HS_;
  v16bf aq0 = load_fragA(qbase, HS_, 0);
  v16bf aq1 = load_fragA(qbase, HS_, 32);

  const unsigned short* kbh = kb + (size_t)bh * T_ * HS_;
  const unsigned short* vbh = vt + (size_t)bh * HS_ * T_;
  unsigned short* Pw = Pl + wave * (16 * 32);

  float mr[8], lr[8];
  v8f o[4] = {};
  #pragma unroll
  for (int i = 0; i < 8; ++i) { mr[i] = -1e30f; lr[i] = 0.0f; }

  const int n_tiles = (q0 + 64) / 32;               // causal bound (block-uniform)

  // issue TDM loads for kv-tile `t` into buffer `bufi`
  auto issue_tile = [&](int t, int bufi) {
    unsigned klds = (unsigned)(uintptr_t)(&Kt[bufi][0]);
    unsigned vlds = (unsigned)(uintptr_t)(&Vt[bufi][0]);
    // K tile: 32 rows (kv) x 64 cols (hs), row pitch 64 elements
    tdm_load_2d_bf16(klds, (unsigned long long)(uintptr_t)(kbh + (size_t)t * 32 * HS_),
                     /*tensor_d0=*/HS_, /*tensor_d1=*/T_,
                     /*tile_d0=*/HS_, /*tile_d1=*/32, /*stride0=*/HS_);
    // V^T tile: 64 rows (hs) x 32 cols (kv), row pitch T elements
    tdm_load_2d_bf16(vlds, (unsigned long long)(uintptr_t)(vbh + (size_t)t * 32),
                     /*tensor_d0=*/T_, /*tensor_d1=*/HS_,
                     /*tile_d0=*/32, /*tile_d1=*/HS_, /*stride0=*/T_);
  };

  if (wave == 0) issue_tile(0, 0);

  for (int it = 0; it < n_tiles; ++it) {
    const int  kv0    = it * 32;
    const int  bufi   = it & 1;
    const bool active = (kv0 <= qw + 15);           // wave-uniform

    if (wave == 0) {
      if (it + 1 < n_tiles) {
        issue_tile(it + 1, bufi ^ 1);               // prefetch next tile via TDM
        __builtin_amdgcn_s_wait_tensorcnt(2);       // current tile's 2 ops done
      } else {
        __builtin_amdgcn_s_wait_tensorcnt(0);
      }
    }
    __syncthreads();                                // K/V tile visible to all waves

    if (active) {
      // S = Q K^T  (16 x 32 per wave, K-dim = HS = 64 -> 2 wmma per column frag)
      v8f s[2] = {};
      #pragma unroll
      for (int n = 0; n < 2; ++n) {
        const unsigned short* kp = &Kt[bufi][(n * 16) * HS_];
        v16bf b0 = load_fragB(kp, HS_, 0);
        v16bf b1 = load_fragB(kp, HS_, 32);
        s[n] = wmma_bf16(aq0, b0, s[n]);
        s[n] = wmma_bf16(aq1, b1, s[n]);
      }
      // causal mask
      #pragma unroll
      for (int n = 0; n < 2; ++n)
        #pragma unroll
        for (int i = 0; i < 8; ++i)
          if (kv0 + n * 16 + cn > qw + i + hi16 * 8) s[n][i] = -1e30f;
      // online softmax per row (row lives across 16-lane half)
      #pragma unroll
      for (int i = 0; i < 8; ++i) {
        float mx = fmaxf(s[0][i], s[1][i]);
        mx = fmaxf(mx, __shfl_xor(mx, 1));
        mx = fmaxf(mx, __shfl_xor(mx, 2));
        mx = fmaxf(mx, __shfl_xor(mx, 4));
        mx = fmaxf(mx, __shfl_xor(mx, 8));
        float mn = fmaxf(mr[i], mx);
        float sc = __expf(mr[i] - mn);
        float p0 = __expf(s[0][i] - mn);
        float p1 = __expf(s[1][i] - mn);
        float rs = p0 + p1;
        rs += __shfl_xor(rs, 1);
        rs += __shfl_xor(rs, 2);
        rs += __shfl_xor(rs, 4);
        rs += __shfl_xor(rs, 8);
        lr[i] = lr[i] * sc + rs;
        mr[i] = mn;
        #pragma unroll
        for (int f = 0; f < 4; ++f) o[f][i] *= sc;
        int prow = i + hi16 * 8;
        Pw[prow * 32 + cn]      = f2bf(p0);
        Pw[prow * 32 + 16 + cn] = f2bf(p1);
      }
    }
    __syncthreads();                                // P visible (wave-local region)
    if (active) {
      // O += P V   (A = P 16x32 from LDS, B = V^T rows from LDS, stride 32)
      v16bf ap = load_fragA(Pw, 32, 0);
      #pragma unroll
      for (int f = 0; f < 4; ++f) {
        v16bf bv = load_fragB(&Vt[bufi][(f * 16) * 32], 32, 0);
        o[f] = wmma_bf16(ap, bv, o[f]);
      }
    }
  }

  // normalize and store O as bf16 in [B, T, H, HS] (ready for output GEMM)
  const int b = bh >> 4, h = bh & 15;
  #pragma unroll
  for (int f = 0; f < 4; ++f)
    #pragma unroll
    for (int i = 0; i < 8; ++i) {
      int row = qw + i + hi16 * 8;
      int hs  = f * 16 + cn;
      float val = o[f][i] / lr[i];
      ob[(((size_t)b * T_ + row) * H_ + h) * HS_ + hs] = f2bf(val);
    }
}

// ---------------------------------------------------------------------------
extern "C" void kernel_launch(void* const* d_in, const int* in_sizes, int n_in,
                              void* d_out, int out_size, void* d_ws, size_t ws_size,
                              hipStream_t stream) {
  const float* x  = (const float*)d_in[0];
  const float* Wq = (const float*)d_in[1];
  const float* Wk = (const float*)d_in[2];
  const float* Wv = (const float*)d_in[3];
  const float* Wp = (const float*)d_in[4];
  const float* bp = (const float*)d_in[5];
  float* out = (float*)d_out;

  const size_t NE = (size_t)B_ * H_ * T_ * HS_;   // 8Mi elements per buffer
  unsigned short* qbuf = (unsigned short*)d_ws;   // Q  [B,H,T,HS] bf16 (scaled)
  unsigned short* kbuf = qbuf + NE;               // K  [B,H,T,HS] bf16
  unsigned short* vtb  = kbuf + NE;               // V^T[B,H,HS,T] bf16
  unsigned short* obuf = vtb + NE;                // O  [B,T,H,HS] bf16

  dim3 blk(128);
  dim3 gp(D_ / 64, M_ / 64);                      // (16, 128)
  proj_kernel<0><<<gp, blk, 0, stream>>>(x, Wq, nullptr, qbuf);
  proj_kernel<1><<<gp, blk, 0, stream>>>(x, Wk, nullptr, kbuf);
  proj_kernel<2><<<gp, blk, 0, stream>>>(x, Wv, nullptr, vtb);

  dim3 ga(T_ / 64, B_ * H_);                      // (32, 64)
  attn_kernel<<<ga, blk, 0, stream>>>(qbuf, kbuf, vtb, obuf);

  proj_kernel<3><<<gp, blk, 0, stream>>>(obuf, Wp, bp, out);

  (void)in_sizes; (void)n_in; (void)out_size; (void)ws_size;
}
